// ToEdge_4879082848654
// MI455X (gfx1250) — matrix operand order repaired and checked by reading, not compiled, not run on hardware
//
#include <hip/hip_runtime.h>

typedef __attribute__((ext_vector_type(16))) __bf16 v16bf;
typedef __attribute__((ext_vector_type(8)))  float  v8f;

union FragB16 {
    v16bf v;
    uint4 q[2];
};

static __device__ __forceinline__ unsigned short f2bf(float f) {
    unsigned int u = __float_as_uint(f);
    unsigned int r = (u + 0x7FFFu + ((u >> 16) & 1u)) >> 16;  // RNE
    return (unsigned short)r;
}
static __device__ __forceinline__ float bf2f(unsigned short u) {
    return __uint_as_float(((unsigned int)u) << 16);
}

#define TILE_E 64
#define XS 392   // 384 + 8 pad (bf16 elems) -> 784B rows: 196 DW, %64 = 4 (bank-rotating)
#define HS 392
#define FS 520   // 512 + 8 pad -> 1040B rows: 260 DW, %64 = 4

// Dynamic LDS layout (bytes)
#define SXF_BYTES (TILE_E * FS * 2)            // 66560: x tile (stride XS) then f tile (stride FS)
#define SH_BYTES  (TILE_E * HS * 2)            // 50176
#define SGATE_OFF (SXF_BYTES + SH_BYTES)       // 116736
#define SMEM_BYTES (SGATE_OFF + TILE_E * 4)    // 116992

// ws layout (bytes)
#define WT1_OFF  0                              // 384*384*2 = 294912
#define WT2_OFF  294912                         // 512*384*2 = 393216
#define WT3_OFF  (294912 + 393216)              // 128*512*2 = 131072
#define W2G_OFF  (WT3_OFF + 131072)             // 384*4     = 1536

// ---------------------------------------------------------------------------
// Prep: transpose+convert weights to bf16 N-major; pack gate column of W2.
// ---------------------------------------------------------------------------
__global__ __launch_bounds__(256) void prep_weights(
    const float* __restrict__ W1, const float* __restrict__ W2,
    const float* __restrict__ W3,
    unsigned short* __restrict__ Wt1, unsigned short* __restrict__ Wt2f,
    unsigned short* __restrict__ Wt3, float* __restrict__ w2g)
{
    int i = blockIdx.x * 256 + threadIdx.x;
    if (i < 384 * 384) {
        int n = i / 384, k = i % 384;
        Wt1[i] = f2bf(W1[k * 384 + n]);
        return;
    }
    int j = i - 384 * 384;
    if (j < 512 * 384) {
        int n = j / 384, k = j % 384;
        Wt2f[j] = f2bf(W2[k * 513 + 1 + n]);
        return;
    }
    int l = j - 512 * 384;
    if (l < 128 * 512) {
        int n = l / 512, k = l % 512;
        Wt3[l] = f2bf(W3[k * 128 + n]);
        return;
    }
    int m = l - 128 * 512;
    if (m < 384) w2g[m] = W2[m * 513];  // gate column, contiguous copy
}

// ---------------------------------------------------------------------------
// Main fused kernel: gather -> GEMM1(relu) -> gate -> GEMM2(*gate) -> GEMM3
// Per n-tile, one wave holds the B fragment and accumulates all 4 m-tiles.
// ---------------------------------------------------------------------------
__global__ __launch_bounds__(256) void edge_mlp_kernel(
    const float* __restrict__ in_vc, const float* __restrict__ ve1,
    const float* __restrict__ ve2, const int* __restrict__ src,
    const int* __restrict__ dst, const float* __restrict__ b1,
    const float* __restrict__ b2, const float* __restrict__ b3,
    const unsigned short* __restrict__ Wt1,
    const unsigned short* __restrict__ Wt2f,
    const unsigned short* __restrict__ Wt3,
    const float* __restrict__ w2g,
    float* __restrict__ out, int E)
{
    extern __shared__ char smem_raw[];
    unsigned short* sXF = (unsigned short*)smem_raw;                 // x then f (aliased)
    unsigned short* sH  = (unsigned short*)(smem_raw + SXF_BYTES);
    float* sGate        = (float*)(smem_raw + SGATE_OFF);

    const int tid = threadIdx.x;
    const int wave = tid >> 5;
    const int lane = tid & 31;
    const int lane16 = lane & 15;
    const int lhalf = lane >> 4;
    const long e0 = (long)blockIdx.x * TILE_E;

    // ---- Phase 1: gather + cvt x tile [TILE_E x 384] into LDS (bf16) ----
    for (int idx = tid; idx < TILE_E * 96; idx += 256) {  // 96 float4 per row
        int e = idx / 96;
        int c4 = (idx % 96) * 4;
        long ge = e0 + e;
        if (ge >= E) ge = E - 1;  // clamp (E % TILE_E == 0 in practice)
        const float* sp;
        if (c4 < 128)      sp = in_vc + (long)src[ge] * 128 + c4;
        else if (c4 < 256) sp = in_vc + (long)dst[ge] * 128 + (c4 - 128);
        else if (c4 < 320) sp = ve1 + ge * 64 + (c4 - 256);
        else               sp = ve2 + ge * 64 + (c4 - 320);
        float4 v = *(const float4*)sp;
        unsigned short* d = &sXF[e * XS + c4];
        d[0] = f2bf(v.x); d[1] = f2bf(v.y); d[2] = f2bf(v.z); d[3] = f2bf(v.w);
    }
    __syncthreads();

    // ---- Phase 2: GEMM1  h = relu(x @ W1 + b1)   [64x384]x[384x384] ----
    for (int nt = wave; nt < 24; nt += 8) {
        v8f acc0 = {}, acc1 = {}, acc2 = {}, acc3 = {};
        const unsigned short* bp = Wt1 + (nt * 16 + lane16) * 384 + 8 * lhalf;
        const unsigned short* ap = &sXF[lane16 * XS + 8 * lhalf];
        for (int kk = 0; kk < 384; kk += 32) {
            FragB16 b, a0, a1, a2, a3;
            b.q[0]  = *(const uint4*)(bp + kk);
            b.q[1]  = *(const uint4*)(bp + kk + 16);
            a0.q[0] = *(const uint4*)(ap + kk);
            a0.q[1] = *(const uint4*)(ap + kk + 16);
            a1.q[0] = *(const uint4*)(ap + 16 * XS + kk);
            a1.q[1] = *(const uint4*)(ap + 16 * XS + kk + 16);
            a2.q[0] = *(const uint4*)(ap + 32 * XS + kk);
            a2.q[1] = *(const uint4*)(ap + 32 * XS + kk + 16);
            a3.q[0] = *(const uint4*)(ap + 48 * XS + kk);
            a3.q[1] = *(const uint4*)(ap + 48 * XS + kk + 16);
            acc0 = __builtin_amdgcn_wmma_f32_16x16x32_bf16(false, a0.v, false, b.v, (short)0, acc0, false, false);
            acc1 = __builtin_amdgcn_wmma_f32_16x16x32_bf16(false, a1.v, false, b.v, (short)0, acc1, false, false);
            acc2 = __builtin_amdgcn_wmma_f32_16x16x32_bf16(false, a2.v, false, b.v, (short)0, acc2, false, false);
            acc3 = __builtin_amdgcn_wmma_f32_16x16x32_bf16(false, a3.v, false, b.v, (short)0, acc3, false, false);
        }
        const int n = nt * 16 + lane16;
        const float bias = b1[n];
#pragma unroll
        for (int r = 0; r < 8; ++r) {
            int m = r + 8 * lhalf;
            float h0 = acc0[r] + bias, h1 = acc1[r] + bias;
            float h2 = acc2[r] + bias, h3 = acc3[r] + bias;
            sH[(m)      * HS + n] = f2bf(h0 > 0.f ? h0 : 0.f);
            sH[(m + 16) * HS + n] = f2bf(h1 > 0.f ? h1 : 0.f);
            sH[(m + 32) * HS + n] = f2bf(h2 > 0.f ? h2 : 0.f);
            sH[(m + 48) * HS + n] = f2bf(h3 > 0.f ? h3 : 0.f);
        }
    }
    __syncthreads();

    // ---- Phase 3: gate_e = sigmoid(h_e . w2g + b2[0]); 4 lanes per edge ----
    {
        const int e = tid >> 2;        // 0..63
        const int part = tid & 3;      // 0..3, within same wave
        const unsigned short* hp = &sH[e * HS + part * 96];
        const float* wg = w2g + part * 96;
        float sum = 0.f;
        for (int k = 0; k < 96; ++k) sum += bf2f(hp[k]) * wg[k];
        sum += __shfl_xor(sum, 1);
        sum += __shfl_xor(sum, 2);
        if (part == 0) sGate[e] = 1.f / (1.f + __expf(-(sum + b2[0])));
    }
    __syncthreads();

    // ---- Phase 4: GEMM2  f = (h @ W2[:,1:] + b2[1:]) * gate -> LDS ----
    for (int nt = wave; nt < 32; nt += 8) {
        v8f acc0 = {}, acc1 = {}, acc2 = {}, acc3 = {};
        const unsigned short* bp = Wt2f + (nt * 16 + lane16) * 384 + 8 * lhalf;
        const unsigned short* ap = &sH[lane16 * HS + 8 * lhalf];
        for (int kk = 0; kk < 384; kk += 32) {
            FragB16 b, a0, a1, a2, a3;
            b.q[0]  = *(const uint4*)(bp + kk);
            b.q[1]  = *(const uint4*)(bp + kk + 16);
            a0.q[0] = *(const uint4*)(ap + kk);
            a0.q[1] = *(const uint4*)(ap + kk + 16);
            a1.q[0] = *(const uint4*)(ap + 16 * HS + kk);
            a1.q[1] = *(const uint4*)(ap + 16 * HS + kk + 16);
            a2.q[0] = *(const uint4*)(ap + 32 * HS + kk);
            a2.q[1] = *(const uint4*)(ap + 32 * HS + kk + 16);
            a3.q[0] = *(const uint4*)(ap + 48 * HS + kk);
            a3.q[1] = *(const uint4*)(ap + 48 * HS + kk + 16);
            acc0 = __builtin_amdgcn_wmma_f32_16x16x32_bf16(false, a0.v, false, b.v, (short)0, acc0, false, false);
            acc1 = __builtin_amdgcn_wmma_f32_16x16x32_bf16(false, a1.v, false, b.v, (short)0, acc1, false, false);
            acc2 = __builtin_amdgcn_wmma_f32_16x16x32_bf16(false, a2.v, false, b.v, (short)0, acc2, false, false);
            acc3 = __builtin_amdgcn_wmma_f32_16x16x32_bf16(false, a3.v, false, b.v, (short)0, acc3, false, false);
        }
        const int j = nt * 16 + lane16;           // f column (= y column j+1)
        const float bias = b2[1 + j];
#pragma unroll
        for (int r = 0; r < 8; ++r) {
            int m = r + 8 * lhalf;
            sXF[(m)      * FS + j] = f2bf((acc0[r] + bias) * sGate[m]);
            sXF[(m + 16) * FS + j] = f2bf((acc1[r] + bias) * sGate[m + 16]);
            sXF[(m + 32) * FS + j] = f2bf((acc2[r] + bias) * sGate[m + 32]);
            sXF[(m + 48) * FS + j] = f2bf((acc3[r] + bias) * sGate[m + 48]);
        }
    }
    __syncthreads();

    // ---- Phase 5: GEMM3  out = f @ W3 + b3   [64x512]x[512x128] ----
    for (int nt = wave; nt < 8; nt += 8) {
        v8f acc0 = {}, acc1 = {}, acc2 = {}, acc3 = {};
        const unsigned short* bp = Wt3 + (nt * 16 + lane16) * 512 + 8 * lhalf;
        const unsigned short* ap = &sXF[lane16 * FS + 8 * lhalf];
        for (int kk = 0; kk < 512; kk += 32) {
            FragB16 b, a0, a1, a2, a3;
            b.q[0]  = *(const uint4*)(bp + kk);
            b.q[1]  = *(const uint4*)(bp + kk + 16);
            a0.q[0] = *(const uint4*)(ap + kk);
            a0.q[1] = *(const uint4*)(ap + kk + 16);
            a1.q[0] = *(const uint4*)(ap + 16 * FS + kk);
            a1.q[1] = *(const uint4*)(ap + 16 * FS + kk + 16);
            a2.q[0] = *(const uint4*)(ap + 32 * FS + kk);
            a2.q[1] = *(const uint4*)(ap + 32 * FS + kk + 16);
            a3.q[0] = *(const uint4*)(ap + 48 * FS + kk);
            a3.q[1] = *(const uint4*)(ap + 48 * FS + kk + 16);
            acc0 = __builtin_amdgcn_wmma_f32_16x16x32_bf16(false, a0.v, false, b.v, (short)0, acc0, false, false);
            acc1 = __builtin_amdgcn_wmma_f32_16x16x32_bf16(false, a1.v, false, b.v, (short)0, acc1, false, false);
            acc2 = __builtin_amdgcn_wmma_f32_16x16x32_bf16(false, a2.v, false, b.v, (short)0, acc2, false, false);
            acc3 = __builtin_amdgcn_wmma_f32_16x16x32_bf16(false, a3.v, false, b.v, (short)0, acc3, false, false);
        }
        const int n = nt * 16 + lane16;
        const float bias = b3[n];
#pragma unroll
        for (int r = 0; r < 8; ++r) {
            long m = r + 8 * lhalf;
            if (e0 + m + 48 < E) {  // full tile (E % 64 == 0 in practice)
                out[(e0 + m)      * 128 + n] = acc0[r] + bias;
                out[(e0 + m + 16) * 128 + n] = acc1[r] + bias;
                out[(e0 + m + 32) * 128 + n] = acc2[r] + bias;
                out[(e0 + m + 48) * 128 + n] = acc3[r] + bias;
            } else {
                if (e0 + m < E)      out[(e0 + m)      * 128 + n] = acc0[r] + bias;
                if (e0 + m + 16 < E) out[(e0 + m + 16) * 128 + n] = acc1[r] + bias;
                if (e0 + m + 32 < E) out[(e0 + m + 32) * 128 + n] = acc2[r] + bias;
                if (e0 + m + 48 < E) out[(e0 + m + 48) * 128 + n] = acc3[r] + bias;
            }
        }
    }
}

// ---------------------------------------------------------------------------
extern "C" void kernel_launch(void* const* d_in, const int* in_sizes, int n_in,
                              void* d_out, int out_size, void* d_ws, size_t ws_size,
                              hipStream_t stream)
{
    const float* in_vc = (const float*)d_in[0];
    const float* ve1   = (const float*)d_in[1];
    const float* ve2   = (const float*)d_in[2];
    const int*   src   = (const int*)d_in[3];
    const int*   dst   = (const int*)d_in[4];
    const float* W1    = (const float*)d_in[5];
    const float* b1    = (const float*)d_in[6];
    const float* W2    = (const float*)d_in[7];
    const float* b2    = (const float*)d_in[8];
    const float* W3    = (const float*)d_in[9];
    const float* b3    = (const float*)d_in[10];
    float* out = (float*)d_out;

    const int E = in_sizes[3];  // number of edges

    char* ws = (char*)d_ws;
    unsigned short* Wt1  = (unsigned short*)(ws + WT1_OFF);
    unsigned short* Wt2f = (unsigned short*)(ws + WT2_OFF);
    unsigned short* Wt3  = (unsigned short*)(ws + WT3_OFF);
    float*          w2g  = (float*)(ws + W2G_OFF);

    // Opt in to >64KB dynamic LDS (gfx1250 WGP has 320KB). Idempotent host call.
    (void)hipFuncSetAttribute((const void*)edge_mlp_kernel,
                              hipFuncAttributeMaxDynamicSharedMemorySize, SMEM_BYTES);

    const int prep_elems = 384 * 384 + 512 * 384 + 128 * 512 + 384;  // 409984
    prep_weights<<<(prep_elems + 255) / 256, 256, 0, stream>>>(W1, W2, W3, Wt1, Wt2f, Wt3, w2g);

    const int nblocks = (E + TILE_E - 1) / TILE_E;  // 12500 for E=800000
    edge_mlp_kernel<<<nblocks, 256, SMEM_BYTES, stream>>>(in_vc, ve1, ve2, src, dst,
                                                          b1, b2, b3,
                                                          Wt1, Wt2f, Wt3, w2g, out, E);
}
